// TransformerBlock_Attention_Chosen_QMLP_10651518894459
// MI455X (gfx1250) — compile-verified, matrix operand order checked by reading, not run on hardware
//
#include <hip/hip_runtime.h>

// MI455X / gfx1250 implementation: wave32, WMMA f16->f32 path.
// Pipeline: transpose+cvt weights -> LN1 -> QKV gemms -> qnorm -> flash attn
//           -> O-proj(+residual) -> LN2 -> w1(771 cols) -> folded VQC+gelu -> w2(+residual)
// v2: DPP16 softmax reductions (no ds_bpermute), 32x64 GEMM wave tiles,
//     64-key attention blocks.

typedef _Float16 half_t;
typedef __attribute__((ext_vector_type(16))) _Float16 v16h;
typedef __attribute__((ext_vector_type(8)))  _Float16 v8h;
typedef __attribute__((ext_vector_type(8)))  float    v8f;

#define TOK   4096   // B*S
#define S_LEN 2048
#define NHEAD 12
#define HD    64
#define EMB   768
#define QM    3072   // Q*M
#define KVQC  800    // padded K for folded vqc gemm (valid 0..770)
#define NVQ1  771    // used columns of hmid

union F8 { v8f v; float f[8]; };

static __device__ __forceinline__ v16h ldfrag(const half_t* p) {
  v8h lo = *(const v8h*)(p);
  v8h hi = *(const v8h*)(p + 16);
  return __builtin_shufflevector(lo, hi, 0,1,2,3,4,5,6,7,8,9,10,11,12,13,14,15);
}

static __device__ __forceinline__ v8f wmma32(v16h a, v16h b, v8f c) {
  return __builtin_amdgcn_wmma_f32_16x16x32_f16(false, a, false, b, (short)0, c, false, false);
}

// ---- DPP16 cross-lane reductions within each 16-lane half (VALU, no LDS) ----
template<int CTRL>
static __device__ __forceinline__ float dppf(float x) {
  return __int_as_float(__builtin_amdgcn_update_dpp(0, __float_as_int(x), CTRL, 0xF, 0xF, true));
}
static __device__ __forceinline__ float redmax16(float x) {
  x = fmaxf(x, dppf<0xB1>(x));   // quad_perm [1,0,3,2]  (xor 1)
  x = fmaxf(x, dppf<0x4E>(x));   // quad_perm [2,3,0,1]  (xor 2)
  x = fmaxf(x, dppf<0x141>(x));  // row_half_mirror      (reduce across octs)
  x = fmaxf(x, dppf<0x140>(x));  // row_mirror           (reduce across 16)
  return x;
}
static __device__ __forceinline__ float redsum16(float x) {
  x += dppf<0xB1>(x);
  x += dppf<0x4E>(x);
  x += dppf<0x141>(x);
  x += dppf<0x140>(x);
  return x;
}

// ---------------- prep kernels ----------------

// WT[n*K + k] = (f16) W[k*N + n]
__global__ __launch_bounds__(256)
void transpose_cvt_kernel(const float* __restrict__ W, half_t* __restrict__ WT, int K, int N)
{
  int idx = blockIdx.x * 256 + threadIdx.x;
  if (idx >= K * N) return;
  int n = idx / K;
  int k = idx - n * K;
  WT[idx] = (half_t)W[(size_t)k * N + n];
}

// wbigT[c*800 + u] = wvqc[(u - (c&3))*768 + (c>>2)] when in range, else 0.
__global__ __launch_bounds__(256)
void build_wbigT_kernel(const float* __restrict__ wvqc, half_t* __restrict__ wbigT)
{
  int idx = blockIdx.x * 256 + threadIdx.x;
  if (idx >= QM * KVQC) return;
  int c = idx / KVQC;
  int u = idx - c * KVQC;
  int i = c & 3, m = c >> 2;
  int j = u - i;
  float v = (j >= 0 && j < 768) ? wvqc[(size_t)j * 768 + m] : 0.f;
  wbigT[idx] = (half_t)v;
}

__global__ __launch_bounds__(256)
void layernorm_f16_kernel(const float* __restrict__ x, const float* __restrict__ g,
                          const float* __restrict__ b, half_t* __restrict__ out)
{
  const int row = blockIdx.x;
  const float* xr = x + (size_t)row * EMB;
  float s1 = 0.f, s2 = 0.f;
  for (int i = threadIdx.x; i < EMB; i += 256) { float v = xr[i]; s1 += v; s2 += v * v; }
  #pragma unroll
  for (int off = 16; off >= 1; off >>= 1) { s1 += __shfl_xor(s1, off, 32); s2 += __shfl_xor(s2, off, 32); }
  __shared__ float sh1[8], sh2[8];
  const int wv = threadIdx.x >> 5;
  if ((threadIdx.x & 31) == 0) { sh1[wv] = s1; sh2[wv] = s2; }
  __syncthreads();
  float a1 = 0.f, a2 = 0.f;
  #pragma unroll
  for (int w = 0; w < 8; ++w) { a1 += sh1[w]; a2 += sh2[w]; }
  const float mean = a1 * (1.f / EMB);
  const float var  = a2 * (1.f / EMB) - mean * mean;
  const float inv  = rsqrtf(var + 1e-5f);
  half_t* orow = out + (size_t)row * EMB;
  for (int i = threadIdx.x; i < EMB; i += 256)
    orow[i] = (half_t)((xr[i] - mean) * inv * g[i] + b[i]);
}

// alpha[r] = 1/sqrt(64 * max(||q_r||^2, 1e-5)) ;  qf layout [b,h,s,d]
__global__ __launch_bounds__(256)
void qnorm_alpha_kernel(const half_t* __restrict__ qf, float* __restrict__ alpha)
{
  int r = blockIdx.x * 256 + threadIdx.x;
  if (r >= 2 * NHEAD * S_LEN) return;
  const half_t* p = qf + (size_t)r * HD;
  float s = 0.f;
  #pragma unroll 8
  for (int i = 0; i < HD; ++i) { float v = (float)p[i]; s += v * v; }
  alpha[r] = rsqrtf((float)HD * fmaxf(s, 1e-5f));
}

// ---------------- generic WMMA gemm: C[M,N] = A[M,K] * Bt[N,K]^T (+epilogue) ----------------
// block = 128 threads = 4 waves; wave tile 32x64 (2 A-frags share each B-frag);
// block tile 128x64.
// MODE 0: Q/K proj -> f16 [b,h,s,d]      MODE 1: V proj -> f16 [b,h,d,s]
// MODE 2: +bias +resid -> f32 [tok,768]  MODE 3: w1 -> f16 hmid[tok,800] (valid<771, zero-pad)
// MODE 4: folded vqc (+bvqc[n>>2]) + exact gelu -> f16 [tok,3072]
template<int MODE>
__global__ __launch_bounds__(128)
void gemm_wmma_kernel(const half_t* __restrict__ A, const half_t* __restrict__ Bt,
                      const float* __restrict__ bias, const float* __restrict__ resid,
                      void* __restrict__ outp, int K)
{
  const int lane = threadIdx.x & 31;
  const int wave = threadIdx.x >> 5;
  const int l16  = lane & 15;
  const int hf   = lane >> 4;
  const int rowbase = blockIdx.y * 128 + wave * 32;
  const int colbase = blockIdx.x * 64;

  const half_t* arow0 = A + (size_t)(rowbase + l16) * K + hf * 8;
  const half_t* arow1 = arow0 + (size_t)16 * K;
  const half_t* brow  = Bt + (size_t)(colbase + l16) * K + hf * 8;
  const size_t bs16 = (size_t)16 * K;

  v8f acc[2][4] = {};
  for (int k0 = 0; k0 < K; k0 += 32) {
    v16h a0 = ldfrag(arow0 + k0);
    v16h a1 = ldfrag(arow1 + k0);
    #pragma unroll
    for (int c = 0; c < 4; ++c) {
      v16h bf = ldfrag(brow + k0 + (size_t)c * bs16);
      acc[0][c] = wmma32(a0, bf, acc[0][c]);
      acc[1][c] = wmma32(a1, bf, acc[1][c]);
    }
  }

  #pragma unroll
  for (int rh = 0; rh < 2; ++rh) {
    F8 r[4];
    #pragma unroll
    for (int c = 0; c < 4; ++c) r[c].v = acc[rh][c];
    #pragma unroll
    for (int c = 0; c < 4; ++c) {
      #pragma unroll
      for (int j = 0; j < 8; ++j) {
        const int row = rowbase + rh * 16 + hf * 8 + j; // token index
        const int n   = colbase + c * 16 + l16;         // output column
        if (MODE == 0) {
          float v = r[c].f[j] + bias[n];
          const int bb = row >> 11, ss = row & (S_LEN - 1);
          const int hh = n >> 6,  dd = n & (HD - 1);
          ((half_t*)outp)[(((size_t)(bb * NHEAD + hh)) * S_LEN + ss) * HD + dd] = (half_t)v;
        } else if (MODE == 1) {
          float v = r[c].f[j] + bias[n];
          const int bb = row >> 11, ss = row & (S_LEN - 1);
          const int hh = n >> 6,  dd = n & (HD - 1);
          ((half_t*)outp)[(((size_t)(bb * NHEAD + hh)) * HD + dd) * S_LEN + ss] = (half_t)v;
        } else if (MODE == 2) {
          float v = r[c].f[j] + bias[n];
          ((float*)outp)[(size_t)row * EMB + n] = v + resid[(size_t)row * EMB + n];
        } else if (MODE == 3) {
          if (n < NVQ1) {
            float v = r[c].f[j] + bias[n];
            ((half_t*)outp)[(size_t)row * KVQC + n] = (half_t)v;
          } else if (n < KVQC) {
            ((half_t*)outp)[(size_t)row * KVQC + n] = (half_t)0.0f;
          }
        } else { // MODE 4
          float v = r[c].f[j] + bias[n >> 2];
          float ge = 0.5f * v * (1.f + erff(v * 0.70710678118654752440f));
          ((half_t*)outp)[(size_t)row * QM + n] = (half_t)ge;
        }
      }
    }
  }
}

// ---------------- flash attention: one wave per (head, 16-query block) ----------------
// Streams keys in blocks of 64; online softmax with DPP16 row reductions.
__global__ __launch_bounds__(128)
void flash_attn_kernel(const half_t* __restrict__ qf, const half_t* __restrict__ kf,
                       const half_t* __restrict__ vt, const float* __restrict__ alpha,
                       half_t* __restrict__ vals)
{
  __shared__ half_t psh[4][16 * 64];
  const int lane = threadIdx.x & 31;
  const int wave = threadIdx.x >> 5;
  const int l16 = lane & 15, hf = lane >> 4;
  const int wid  = blockIdx.x * 4 + wave;
  const int bh   = wid >> 7;    // S/16 = 128 query blocks per head
  const int qblk = wid & 127;

  const half_t* qb = qf + ((size_t)bh * S_LEN + qblk * 16) * HD;
  const v16h q0 = ldfrag(qb + l16 * HD + hf * 8);
  const v16h q1 = ldfrag(qb + l16 * HD + hf * 8 + 32);

  float al[8], mst[8], lst[8];
  #pragma unroll
  for (int j = 0; j < 8; ++j) {
    al[j]  = alpha[(size_t)bh * S_LEN + qblk * 16 + hf * 8 + j];
    mst[j] = -1e30f;
    lst[j] = 0.f;
  }
  v8f o[4] = {};
  const half_t* kb = kf + (size_t)bh * S_LEN * HD;
  const half_t* vb = vt + (size_t)bh * HD * S_LEN;
  half_t* pw = psh[wave];

  for (int t0 = 0; t0 < S_LEN; t0 += 64) {
    // scores: S[16 x 64] = Q(16x64) @ K_blk^T
    v8f s[4] = {};
    #pragma unroll
    for (int nt = 0; nt < 4; ++nt) {
      const half_t* kp = kb + (size_t)(t0 + nt * 16 + l16) * HD + hf * 8;
      s[nt] = wmma32(q0, ldfrag(kp),      s[nt]);
      s[nt] = wmma32(q1, ldfrag(kp + 32), s[nt]);
    }

    F8 u[4];
    #pragma unroll
    for (int nt = 0; nt < 4; ++nt) u[nt].v = s[nt];

    float corr[8];
    #pragma unroll
    for (int j = 0; j < 8; ++j) {
      const float a  = al[j];
      const float x0 = u[0].f[j] * a;
      const float x1 = u[1].f[j] * a;
      const float x2 = u[2].f[j] * a;
      const float x3 = u[3].f[j] * a;
      float mx = redmax16(fmaxf(fmaxf(x0, x1), fmaxf(x2, x3)));
      const float nm = fmaxf(mst[j], mx);
      const float p0 = __expf(x0 - nm);
      const float p1 = __expf(x1 - nm);
      const float p2 = __expf(x2 - nm);
      const float p3 = __expf(x3 - nm);
      const float rs = redsum16((p0 + p1) + (p2 + p3));
      corr[j] = __expf(mst[j] - nm);
      lst[j]  = lst[j] * corr[j] + rs;
      mst[j]  = nm;
      half_t* pr = pw + (hf * 8 + j) * 64 + l16;
      pr[0]  = (half_t)p0;
      pr[16] = (half_t)p1;
      pr[32] = (half_t)p2;
      pr[48] = (half_t)p3;
    }
    { // rescale running output accumulators
      #pragma unroll
      for (int nd = 0; nd < 4; ++nd) {
        F8 t; t.v = o[nd];
        #pragma unroll
        for (int j = 0; j < 8; ++j) t.f[j] *= corr[j];
        o[nd] = t.v;
      }
    }
    __builtin_amdgcn_wave_barrier();  // wave-private LDS; DS is in-order within a wave
    const v16h pf0 = ldfrag(pw + l16 * 64 + hf * 8);       // keys 0..31 of block
    const v16h pf1 = ldfrag(pw + l16 * 64 + 32 + hf * 8);  // keys 32..63 of block
    const half_t* vp = vb + (size_t)l16 * S_LEN + t0 + hf * 8;
    #pragma unroll
    for (int nd = 0; nd < 4; ++nd) {
      const half_t* vpt = vp + (size_t)nd * 16 * S_LEN;
      o[nd] = wmma32(pf0, ldfrag(vpt),      o[nd]);
      o[nd] = wmma32(pf1, ldfrag(vpt + 32), o[nd]);
    }
  }

  const int bb = bh / NHEAD, hh = bh % NHEAD;
  #pragma unroll
  for (int c = 0; c < 4; ++c) {
    F8 r; r.v = o[c];
    #pragma unroll
    for (int j = 0; j < 8; ++j) {
      const int ss = qblk * 16 + hf * 8 + j;
      const int dd = c * 16 + l16;
      vals[((size_t)(bb * S_LEN + ss)) * EMB + hh * HD + dd] = (half_t)(r.f[j] / lst[j]);
    }
  }
}

// ---------------- launcher ----------------
extern "C" void kernel_launch(void* const* d_in, const int* in_sizes, int n_in,
                              void* d_out, int out_size, void* d_ws, size_t ws_size,
                              hipStream_t stream)
{
  (void)in_sizes; (void)n_in; (void)out_size; (void)ws_size;

  const float* x    = (const float*)d_in[0];
  const float* ln1g = (const float*)d_in[1];
  const float* ln1b = (const float*)d_in[2];
  const float* wq   = (const float*)d_in[3];
  const float* bq   = (const float*)d_in[4];
  const float* wk   = (const float*)d_in[5];
  const float* bk   = (const float*)d_in[6];
  const float* wvw  = (const float*)d_in[7];
  const float* bv   = (const float*)d_in[8];
  const float* wo   = (const float*)d_in[9];
  const float* bo   = (const float*)d_in[10];
  const float* ln2g = (const float*)d_in[11];
  const float* ln2b = (const float*)d_in[12];
  const float* w1   = (const float*)d_in[13];
  const float* b1   = (const float*)d_in[14];
  const float* wvqc = (const float*)d_in[15];
  const float* bvqc = (const float*)d_in[16];
  const float* w2   = (const float*)d_in[17];
  const float* b2   = (const float*)d_in[18];

  char* ws = (char*)d_ws;
  size_t off = 0;
  auto alloc = [&](size_t bytes) -> void* {
    void* p = ws + off;
    off = (off + bytes + 255) & ~(size_t)255;
    return p;
  };

  half_t* xn1    = (half_t*)alloc((size_t)TOK * EMB * 2);
  half_t* wqT    = (half_t*)alloc((size_t)EMB * EMB * 2);
  half_t* wkT    = (half_t*)alloc((size_t)EMB * EMB * 2);
  half_t* wvT    = (half_t*)alloc((size_t)EMB * EMB * 2);
  half_t* woT    = (half_t*)alloc((size_t)EMB * EMB * 2);
  half_t* w1T    = (half_t*)alloc((size_t)QM * EMB * 2);
  half_t* w2T    = (half_t*)alloc((size_t)EMB * QM * 2);
  half_t* wbigT  = (half_t*)alloc((size_t)QM * KVQC * 2);
  half_t* qfh    = (half_t*)alloc((size_t)TOK * EMB * 2);
  half_t* kfh    = (half_t*)alloc((size_t)TOK * EMB * 2);
  half_t* vth    = (half_t*)alloc((size_t)TOK * EMB * 2);
  float*  alphaf = (float*) alloc((size_t)2 * NHEAD * S_LEN * 4);
  half_t* valsh  = (half_t*)alloc((size_t)TOK * EMB * 2);
  float*  xresf  = (float*) alloc((size_t)TOK * EMB * 4);
  half_t* xn2    = (half_t*)alloc((size_t)TOK * EMB * 2);
  half_t* hmidh  = (half_t*)alloc((size_t)TOK * KVQC * 2);
  half_t* ghvh   = (half_t*)alloc((size_t)TOK * QM * 2);

  // weight prep: f32 [K,N] -> f16 [N,K]
  transpose_cvt_kernel<<<(EMB*EMB + 255)/256, 256, 0, stream>>>(wq,  wqT, EMB, EMB);
  transpose_cvt_kernel<<<(EMB*EMB + 255)/256, 256, 0, stream>>>(wk,  wkT, EMB, EMB);
  transpose_cvt_kernel<<<(EMB*EMB + 255)/256, 256, 0, stream>>>(wvw, wvT, EMB, EMB);
  transpose_cvt_kernel<<<(EMB*EMB + 255)/256, 256, 0, stream>>>(wo,  woT, EMB, EMB);
  transpose_cvt_kernel<<<(EMB*QM  + 255)/256, 256, 0, stream>>>(w1,  w1T, EMB, QM);
  transpose_cvt_kernel<<<(QM*EMB  + 255)/256, 256, 0, stream>>>(w2,  w2T, QM, EMB);
  build_wbigT_kernel<<<(QM*KVQC + 255)/256, 256, 0, stream>>>(wvqc, wbigT);

  layernorm_f16_kernel<<<TOK, 256, 0, stream>>>(x, ln1g, ln1b, xn1);

  dim3 g12(12, 32), g13(13, 32), g48(48, 32);
  gemm_wmma_kernel<0><<<g12, 128, 0, stream>>>(xn1, wqT, bq, nullptr, qfh, EMB);
  gemm_wmma_kernel<0><<<g12, 128, 0, stream>>>(xn1, wkT, bk, nullptr, kfh, EMB);
  gemm_wmma_kernel<1><<<g12, 128, 0, stream>>>(xn1, wvT, bv, nullptr, vth, EMB);

  qnorm_alpha_kernel<<<(2*NHEAD*S_LEN + 255)/256, 256, 0, stream>>>(qfh, alphaf);
  flash_attn_kernel<<<(2*NHEAD*(S_LEN/16))/4, 128, 0, stream>>>(qfh, kfh, vth, alphaf, valsh);

  gemm_wmma_kernel<2><<<g12, 128, 0, stream>>>(valsh, woT, bo, x, xresf, EMB);
  layernorm_f16_kernel<<<TOK, 256, 0, stream>>>(xresf, ln2g, ln2b, xn2);
  gemm_wmma_kernel<3><<<g13, 128, 0, stream>>>(xn2, w1T, b1, nullptr, hmidh, EMB);
  gemm_wmma_kernel<4><<<g48, 128, 0, stream>>>(hmidh, wbigT, bvqc, nullptr, ghvh, KVQC);
  gemm_wmma_kernel<2><<<g12, 128, 0, stream>>>(ghvh, w2T, b2, xresf, d_out, QM);
}
